// IterativeARLayer_73641509257887
// MI455X (gfx1250) — matrix operand order repaired and checked by reading, not compiled, not run on hardware
//
#include <hip/hip_runtime.h>

#define B_N  16384
#define D_N  64
#define R_N  63
#define H_N  256
#define PK_N 24

#define W1S     72    // LDS row stride (bf16) of W1 panel (64 + pad) -> distinct banks per row-lane
#define W2S     264   // LDS row stride (bf16) of W2 panel (256 + pad)
#define HSTRIDE 72    // staged h row stride (bf16)
#define RSTRIDE 28    // staged raw row stride (f32), 112B rows -> 16B-aligned float4 reads
#define PI_F 3.14159265358979323846f

// dynamic LDS layout (bytes)
#define W1P_SZ  (2 * 256 * W1S * 2)       // 73728  : double-buffered W1 panel
#define W2P_SZ  (2 * 32 * W2S * 2)        // 33792  : double-buffered W2 panel
#define HS_SZ   (4 * 32 * HSTRIDE * 2)    // 18432  : per-wave h staging (32 rows)
#define RS_SZ   (4 * 32 * RSTRIDE * 4)    // 14336  : per-wave raw staging
#define W2P_OFF (W1P_SZ)
#define HS_OFF  (W2P_OFF + W2P_SZ)
#define RS_OFF  (HS_OFF + HS_SZ)
#define SMEM_BYTES (RS_OFF + RS_SZ)       // 140288

typedef __bf16 bf16_t;
typedef __attribute__((ext_vector_type(16))) __bf16 v16bf;
typedef __attribute__((ext_vector_type(8)))  __bf16 v8bf;
typedef __attribute__((ext_vector_type(4)))  __bf16 v4bf;
typedef __attribute__((ext_vector_type(8)))  float  v8f;

#if __has_builtin(__builtin_amdgcn_global_load_async_to_lds_b128)
#define HAVE_ASYNC 1
#else
#define HAVE_ASYNC 0
#endif

// 16-byte int vector matching the async builtin's pointee type exactly
typedef int v4i_vs __attribute__((vector_size(4 * sizeof(int))));
typedef v4i_vs __attribute__((address_space(1)))* gptr_v4i;   // global
typedef v4i_vs __attribute__((address_space(3)))* lptr_v4i;   // LDS

// Copy one 16B chunk global->LDS (async DMA on gfx1250, ASYNCcnt-tracked).
static __device__ __forceinline__ void copy16(const bf16_t* g, bf16_t* l) {
#if HAVE_ASYNC
    __builtin_amdgcn_global_load_async_to_lds_b128(
        (gptr_v4i)(void*)g,
        (lptr_v4i)(void*)l, 0, 0);
#else
    *(v8bf*)l = *(const v8bf*)g;
#endif
}

static __device__ __forceinline__ void wait_async_fence() {
#if HAVE_ASYNC
#if __has_builtin(__builtin_amdgcn_s_wait_asynccnt)
    __builtin_amdgcn_s_wait_asynccnt(0);
#else
    asm volatile("s_wait_asynccnt 0" ::: "memory");
#endif
#endif
}

// A-operand (16x32 bf16): lane<16 holds K {0..7,16..23}, lane>=16 holds K {8..15,24..31}.
// Caller pre-offsets base by half*8.
static __device__ __forceinline__ v16bf ld_a(const bf16_t* p) {
    v8bf lo = *(const v8bf*)(p);
    v8bf hi = *(const v8bf*)(p + 16);
    return __builtin_shufflevector(lo, hi, 0,1,2,3,4,5,6,7,8,9,10,11,12,13,14,15);
}
// B-operand (32x16 bf16): lane<16 holds K {0..7,16..23}, lane>=16 holds K {8..15,24..31}; n = lane%16.
// Caller pre-offsets base by half*8.
static __device__ __forceinline__ v16bf ld_b(const bf16_t* p) {
    v8bf lo = *(const v8bf*)(p);
    v8bf hi = *(const v8bf*)(p + 16);
    return __builtin_shufflevector(lo, hi, 0,1,2,3,4,5,6,7,8,9,10,11,12,13,14,15);
}

static __device__ __forceinline__ v8f wmma_bf16(v16bf a, v16bf b, v8f c) {
    return __builtin_amdgcn_wmma_f32_16x16x32_bf16(false, a, false, b, (short)0, c, false, false);
}

// ---------------- prep kernels ----------------

__global__ void prep_zb_kernel(const float* __restrict__ z, bf16_t* __restrict__ zb, int n) {
    int i = blockIdx.x * blockDim.x + threadIdx.x;
    if (i < n) zb[i] = (bf16_t)z[i];
}

// W1t[r][h][d] = (d<=r) ? W1[r][d][h] : 0   (causal mask baked in, transposed for B loads)
__global__ void prep_w1_kernel(const float* __restrict__ W1, bf16_t* __restrict__ W1t) {
    int i = blockIdx.x * blockDim.x + threadIdx.x;
    if (i >= R_N * H_N * D_N) return;
    int d = i % D_N;
    int h = (i / D_N) % H_N;
    int r = i / (D_N * H_N);
    float v = (d <= r) ? W1[((size_t)r * D_N + d) * H_N + h] : 0.0f;
    W1t[i] = (bf16_t)v;
}

// W2t[r][p(pad 32)][k] with the h index PERMUTED within each 64-block to match the
// packed h-staging order: stored position k <-> original column (k%4)*16 + k/4.
__global__ void prep_w2_kernel(const float* __restrict__ W2, bf16_t* __restrict__ W2t) {
    int i = blockIdx.x * blockDim.x + threadIdx.x;
    if (i >= R_N * 32 * H_N) return;
    int k = i % H_N;
    int p = (i / H_N) % 32;
    int r = i / (32 * H_N);
    int c  = k >> 6;
    int kk = k & 63;
    int h  = c * 64 + (kk & 3) * 16 + (kk >> 2);   // original h column for stored slot k
    float v = (p < PK_N) ? W2[((size_t)r * H_N + h) * PK_N + p] : 0.0f;
    W2t[i] = (bf16_t)v;
}

// ---------------- panel copy (whole 128-thread block cooperates) ----------------

static __device__ __forceinline__ void load_panels(const bf16_t* __restrict__ w1src,
                                                   const bf16_t* __restrict__ w2src,
                                                   bf16_t* w1dst, bf16_t* w2dst,
                                                   int t, bool fullK) {
    // W1 panel: 256 rows x 64 bf16 -> LDS stride W1S. 2048 16B-chunks, 16 per thread.
#pragma unroll
    for (int j = 0; j < 16; ++j) {
        int q = t + 128 * j;
        int row = q >> 3, cc = q & 7;
        if (fullK || cc < 4)   // d 32..63 all-zero (mask) when r < 32
            copy16(w1src + row * 64 + cc * 8, w1dst + row * W1S + cc * 8);
    }
    // W2 panel: 32 rows x 256 bf16 -> LDS stride W2S. 1024 chunks, 8 per thread.
#pragma unroll
    for (int j = 0; j < 8; ++j) {
        int q = t + 128 * j;
        int row = q >> 5, cc = q & 31;
        copy16(w2src + row * 256 + cc * 8, w2dst + row * W2S + cc * 8);
    }
}

// ---------------- fused main kernel ----------------
// block = 128 threads (4 waves); each wave owns 32 batch rows (2 M-tiles); block = 128 rows.

__global__ void __launch_bounds__(128)
flow_main_kernel(const float* __restrict__ zf,
                 const float* __restrict__ b1,
                 const float* __restrict__ b2,
                 const bf16_t* __restrict__ zb,
                 const bf16_t* __restrict__ W1t,
                 const bf16_t* __restrict__ W2t,
                 float* __restrict__ out_x,
                 float* __restrict__ out_ld) {
    extern __shared__ char smem[];
    bf16_t* w1p   = (bf16_t*)(smem);
    bf16_t* w2p   = (bf16_t*)(smem + W2P_OFF);
    bf16_t* hsall = (bf16_t*)(smem + HS_OFF);
    float*  rsall = (float*)(smem + RS_OFF);

    const int t    = threadIdx.x;
    const int wave = t >> 5;
    const int lane = t & 31;
    const int half = lane >> 4;
    const int ln   = lane & 15;

    bf16_t* hs = hsall + wave * 32 * HSTRIDE;
    float*  rs = rsall + wave * 32 * RSTRIDE;

    const int rowBase = blockIdx.x * 128 + wave * 32;

    // z A-operands, invariant over r: av[mt][kb]
    v16bf av[2][2];
#pragma unroll
    for (int mt = 0; mt < 2; ++mt) {
        const bf16_t* zrow = zb + (size_t)(rowBase + mt * 16 + ln) * D_N;
        av[mt][0] = ld_a(zrow + 0  + half * 8);
        av[mt][1] = ld_a(zrow + 32 + half * 8);
    }

    float ld_acc = 0.0f;
    out_x[(size_t)(rowBase + lane) * D_N + 0] = 0.0f;

    // preload panel for r = 0
    load_panels(W1t, W2t, w1p, w2p, t, /*fullK=*/false);
    wait_async_fence();
    __syncthreads();

#pragma unroll 1
    for (int r = 0; r < R_N; ++r) {
        const int cur = r & 1;
        // kick off async fill of the other buffer for r+1 (overlapped with compute)
        if (r + 1 < R_N)
            load_panels(W1t + (size_t)(r + 1) * H_N * D_N,
                        W2t + (size_t)(r + 1) * 32 * H_N,
                        w1p + (cur ^ 1) * (256 * W1S),
                        w2p + (cur ^ 1) * (32 * W2S),
                        t, (r + 1) >= 32);

        const bf16_t* w1L = w1p + cur * (256 * W1S);
        const bf16_t* w2L = w2p + cur * (32 * W2S);

        // GEMM2 accumulators (2 M-tiles x 2 N-tiles), seeded with b2
        v8f racc[2][2];
#pragma unroll
        for (int nt2 = 0; nt2 < 2; ++nt2) {
            int p = nt2 * 16 + ln;
            float bv = (p < PK_N) ? b2[r * PK_N + p] : 0.0f;
#pragma unroll
            for (int i = 0; i < 8; ++i) { racc[0][nt2][i] = bv; racc[1][nt2][i] = bv; }
        }

#pragma unroll 1
        for (int c = 0; c < 4; ++c) {       // 64 h-columns per chunk
            v8f acc[2][4];
#pragma unroll
            for (int nt = 0; nt < 4; ++nt) {
                float bv = b1[r * H_N + c * 64 + nt * 16 + ln];
#pragma unroll
                for (int i = 0; i < 8; ++i) { acc[0][nt][i] = bv; acc[1][nt][i] = bv; }
            }
            // K-block 0 (d 0..31): one B load feeds both M-tiles
#pragma unroll
            for (int nt = 0; nt < 4; ++nt) {
                v16bf bv = ld_b(w1L + (size_t)(c * 64 + nt * 16 + ln) * W1S + half * 8);
                acc[0][nt] = wmma_bf16(av[0][0], bv, acc[0][nt]);
                acc[1][nt] = wmma_bf16(av[1][0], bv, acc[1][nt]);
            }
            // K-block 1 (d 32..63): masked to zero when r < 32
            if (r >= 32) {
#pragma unroll
                for (int nt = 0; nt < 4; ++nt) {
                    v16bf bv = ld_b(w1L + (size_t)(c * 64 + nt * 16 + ln) * W1S + 32 + half * 8);
                    acc[0][nt] = wmma_bf16(av[0][1], bv, acc[0][nt]);
                    acc[1][nt] = wmma_bf16(av[1][1], bv, acc[1][nt]);
                }
            }
            // relu -> bf16 -> LDS, PACKED: 4 nt-values per lane-row are contiguous
            // (staged column = ln*4 + nt; W2t prep applies the matching permutation)
#pragma unroll
            for (int mt = 0; mt < 2; ++mt)
#pragma unroll
                for (int i = 0; i < 8; ++i) {
                    v4bf pk;
#pragma unroll
                    for (int nt = 0; nt < 4; ++nt) {
                        float v = acc[mt][nt][i];
                        v = v > 0.0f ? v : 0.0f;
                        pk[nt] = (bf16_t)v;
                    }
                    *(v4bf*)(hs + (mt * 16 + i + 8 * half) * HSTRIDE + ln * 4) = pk;
                }
            asm volatile("s_wait_dscnt 0" ::: "memory");   // wave-local LDS transpose handoff

            // reload h chunk in A layout, both M-tiles, two K=32 blocks each
            v16bf ha[2][2];
#pragma unroll
            for (int mt = 0; mt < 2; ++mt) {
                const bf16_t* hrow = hs + (mt * 16 + ln) * HSTRIDE;
                ha[mt][0] = ld_a(hrow + 0  + half * 8);
                ha[mt][1] = ld_a(hrow + 32 + half * 8);
            }
            // GEMM2 partial over this 64-wide K slice
#pragma unroll
            for (int nt2 = 0; nt2 < 2; ++nt2) {
                const bf16_t* wp = w2L + (size_t)(nt2 * 16 + ln) * W2S + c * 64 + half * 8;
                v16bf b0 = ld_b(wp);
                v16bf b1v = ld_b(wp + 32);
#pragma unroll
                for (int mt = 0; mt < 2; ++mt) {
                    racc[mt][nt2] = wmma_bf16(ha[mt][0], b0,  racc[mt][nt2]);
                    racc[mt][nt2] = wmma_bf16(ha[mt][1], b1v, racc[mt][nt2]);
                }
            }
        }

        // raw (32x24 per wave) -> LDS
#pragma unroll
        for (int mt = 0; mt < 2; ++mt)
#pragma unroll
            for (int nt2 = 0; nt2 < 2; ++nt2) {
                int p = nt2 * 16 + ln;
                if (p < PK_N) {
#pragma unroll
                    for (int i = 0; i < 8; ++i)
                        rs[(mt * 16 + i + 8 * half) * RSTRIDE + p] = racc[mt][nt2][i];
                }
            }
        asm volatile("s_wait_dscnt 0" ::: "memory");

        // transcendental epilogue: all 32 lanes, one batch row each
        {
            const int row = rowBase + lane;
            float zi = zf[(size_t)row * D_N + (r + 1)];
            float tt = __tanf((zi - PI_F) * 0.5f);
            const float* rr = rs + lane * RSTRIDE;

            // vectorized raw read: 6x ds_load_b128
            float rawv[24];
#pragma unroll
            for (int k = 0; k < 24; k += 4) {
                float4 qq = *(const float4*)(rr + k);
                rawv[k] = qq.x; rawv[k + 1] = qq.y; rawv[k + 2] = qq.z; rawv[k + 3] = qq.w;
            }

            float ru[8], mx = -1e30f;
#pragma unroll
            for (int k = 0; k < 8; ++k) { ru[k] = rawv[16 + k]; mx = fmaxf(mx, ru[k]); }
            float den = 0.0f;
#pragma unroll
            for (int k = 0; k < 8; ++k) { ru[k] = __expf(ru[k] - mx); den += ru[k]; }
            float inv = __builtin_amdgcn_rcpf(den);
            float t2p1 = 1.0f + tt * tt;
            float xt = 0.0f, dsum = 0.0f;
#pragma unroll
            for (int k = 0; k < 8; ++k) {
                float rho   = ru[k] * inv;
                float alpha = __expf(rawv[k]);
                float g     = alpha * tt + rawv[8 + k];
                xt   += rho * (2.0f * atanf(g) + PI_F);
                dsum += rho * alpha * t2p1 * __builtin_amdgcn_rcpf(1.0f + g * g);
            }
            out_x[(size_t)row * D_N + (r + 1)] = xt;
            ld_acc += __logf(dsum);
        }

        wait_async_fence();   // next panel resident before anyone reads it
        __syncthreads();      // and everyone done with the buffer we'll overwrite next
    }

    out_ld[rowBase + lane] = ld_acc;
}

// ---------------- host launcher ----------------

extern "C" void kernel_launch(void* const* d_in, const int* in_sizes, int n_in,
                              void* d_out, int out_size, void* d_ws, size_t ws_size,
                              hipStream_t stream) {
    const float* z  = (const float*)d_in[0];
    const float* W1 = (const float*)d_in[1];
    const float* b1 = (const float*)d_in[2];
    const float* W2 = (const float*)d_in[3];
    const float* b2 = (const float*)d_in[4];

    char* ws = (char*)d_ws;
    bf16_t* zb  = (bf16_t*)(ws);
    bf16_t* W1t = (bf16_t*)(ws + (size_t)B_N * D_N * 2);
    bf16_t* W2t = (bf16_t*)(ws + (size_t)B_N * D_N * 2 + (size_t)R_N * H_N * D_N * 2);

    float* out_x  = (float*)d_out;
    float* out_ld = (float*)d_out + (size_t)B_N * D_N;

    {
        int n = B_N * D_N;
        prep_zb_kernel<<<(n + 255) / 256, 256, 0, stream>>>(z, zb, n);
    }
    {
        int n = R_N * H_N * D_N;
        prep_w1_kernel<<<(n + 255) / 256, 256, 0, stream>>>(W1, W1t);
    }
    {
        int n = R_N * 32 * H_N;
        prep_w2_kernel<<<(n + 255) / 256, 256, 0, stream>>>(W2, W2t);
    }

    // 32 rows per wave, 4 waves per block -> 128 rows per block
    flow_main_kernel<<<B_N / 128, 128, SMEM_BYTES, stream>>>(
        z, b1, b2, zb, W1t, W2t, out_x, out_ld);
}
